// AttentionModel_76149770158120
// MI455X (gfx1250) — matrix operand order repaired and checked by reading, not compiled, not run on hardware
//
#include <hip/hip_runtime.h>
#include <hip/hip_fp16.h>

typedef __attribute__((ext_vector_type(16))) _Float16 v16h;
typedef __attribute__((ext_vector_type(8)))  _Float16 v8h;
typedef __attribute__((ext_vector_type(8)))  float    v8f;
typedef __attribute__((ext_vector_type(4)))  int      v4i;

#define B_ROWS 262144
#define C_DIM  256
#define P_DIM  100
#define P_PAD  112
#define NT     7       // N tiles of 16 (covers 112 >= 100)

// Packed u/v fragment image: [khalf(2)][matrix(2)][t(7)][ksl(4)][half(2)][lane(32)] x 16B
#define PIECES_TOTAL    (2 * 2 * NT * 4 * 2 * 32)       // 7168 pieces of 16B
#define KHALF_BYTES     (2 * NT * 4 * 2 * 32 * 16)      // 57344 B per K-half (U+V)
#define MAT_BYTES       (NT * 4 * 2 * 32 * 16)          // 28672 B per matrix per K-half

// ---------------------------------------------------------------------------
// Prep: build the f16 fragment-swizzled image of u and v (zero-padded to 112
// rows) so the GEMM kernel can stage it to LDS with straight 16B copies and
// read fragments as conflict-free ds_load_b128 at (base + lane*16).
// Piece p -> bytes [p*16, p*16+16) with
//   p = ((((khalf*2 + m)*7 + t)*4 + ksl)*2 + half)*32 + lane
// content: mat[row= t*16+(lane&15)][k = (khalf*4+ksl)*32 + (lane>=16?8:0) + half*16 .. +8)
// ---------------------------------------------------------------------------
__global__ __launch_bounds__(256) void prep_kernel(
    const float* __restrict__ u, const float* __restrict__ v,
    const float* __restrict__ w,
    _Float16* __restrict__ uvpack, float* __restrict__ wpad)
{
  int p = blockIdx.x * 256 + threadIdx.x;               // 28 blocks -> 7168 pieces
  if (p < PIECES_TOTAL) {
    int lane  = p & 31;
    int half  = (p >> 5) & 1;
    int ksl   = (p >> 6) & 3;
    int t     = (p >> 8) % NT;
    int m     = (p >> 8) / NT % 2;
    int khalf = (p >> 8) / (NT * 2);

    int row = t * 16 + (lane & 15);
    int k0  = (khalf * 4 + ksl) * 32 + ((lane >> 4) ? 8 : 0) + half * 16;
    const float* src = (m == 0) ? u : v;

    _Float16* dst = uvpack + (size_t)p * 8;             // 8 halves = 16B
#pragma unroll
    for (int j = 0; j < 8; ++j) {
      float f = (row < P_DIM) ? src[row * C_DIM + k0 + j] : 0.0f;
      dst[j] = (_Float16)f;
    }
  }
  if (p < P_PAD) wpad[p] = (p < P_DIM) ? w[p] : 0.0f;
}

// ---------------------------------------------------------------------------
// Fused: alpha[b] = sum_p tanh(feat.u^T) * sigmoid(feat.v^T) * w[p]
// Block = 128 rows (8 waves x 16-row M-tile). u/v fragments staged to LDS in
// two K-halves (56KB each) via async global->LDS copies.
// ---------------------------------------------------------------------------
__global__ __launch_bounds__(256, 1) void fused_alpha_kernel(
    const float*    __restrict__ feat,
    const _Float16* __restrict__ uvpack,
    const float*    __restrict__ wpad,
    float*          __restrict__ alpha)
{
  __shared__ alignas(16) unsigned char smem[KHALF_BYTES];   // 57344 B

  const int tid     = threadIdx.x;
  const int lane    = tid & 31;
  const int wave    = tid >> 5;
  const int lm      = lane & 15;
  const int halfSel = lane >> 4;                 // 0: lanes 0-15, 1: lanes 16-31
  const int row0    = blockIdx.x * 128 + wave * 16;

  // A row pointer; per ISA 16-bit A 16x32 layout the K-chunks per lane are
  //   lanes 0-15 : K = [0..7]  and [16..23]
  //   lanes 16-31: K = [8..15] and [24..31]
  const float* aptr = feat + (size_t)(row0 + lm) * C_DIM + (halfSel ? 8 : 0);

  v8f accU[NT], accV[NT];
#pragma unroll
  for (int t = 0; t < NT; ++t) { accU[t] = v8f{}; accV[t] = v8f{}; }

  for (int kh = 0; kh < 2; ++kh) {
    if (kh) __syncthreads();                     // previous half fully consumed

    // ---- stage 56KB (U+V fragments for this K-half) into LDS ----
    {
      const unsigned char* src = (const unsigned char*)uvpack + (size_t)kh * KHALF_BYTES;
      for (int i = tid; i < KHALF_BYTES / 16; i += 256) {
#if __has_builtin(__builtin_amdgcn_global_load_async_to_lds_b128)
        __builtin_amdgcn_global_load_async_to_lds_b128(
            (__attribute__((address_space(1))) v4i*)(src + (size_t)i * 16),
            (__attribute__((address_space(3))) v4i*)(smem + (size_t)i * 16),
            0, 0);
#else
        *(v8h*)(smem + (size_t)i * 16) = *(const v8h*)(src + (size_t)i * 16);
#endif
      }
#if __has_builtin(__builtin_amdgcn_global_load_async_to_lds_b128)
#if __has_builtin(__builtin_amdgcn_s_wait_asynccnt)
      __builtin_amdgcn_s_wait_asynccnt(0);
#else
      asm volatile("s_wait_asynccnt 0x0" ::: "memory");
#endif
#endif
    }
    __syncthreads();

#pragma unroll 2
    for (int ksl = 0; ksl < 4; ++ksl) {
      const int ks = kh * 128 + ksl * 32;

      // ---- load A tile (fp32, streaming HBM traffic) and convert to f16 ----
      union { float4 q[4]; float s[16]; } A;
      A.q[0] = *(const float4*)(aptr + ks + 0);
      A.q[1] = *(const float4*)(aptr + ks + 4);
      A.q[2] = *(const float4*)(aptr + ks + 16);
      A.q[3] = *(const float4*)(aptr + ks + 20);
      v16h a;
#pragma unroll
      for (int i = 0; i < 16; ++i) a[i] = (_Float16)A.s[i];

      // ---- B fragments from LDS: base + lane*16, conflict-free b128 ----
#pragma unroll
      for (int t = 0; t < NT; ++t) {
        const int fb = (t * 4 + ksl) * 1024 + lane * 16;
        union { v8h h[2]; v16h v; } BU, BV;
        BU.h[0] = *(const v8h*)(smem + fb);
        BU.h[1] = *(const v8h*)(smem + fb + 512);
        BV.h[0] = *(const v8h*)(smem + MAT_BYTES + fb);
        BV.h[1] = *(const v8h*)(smem + MAT_BYTES + fb + 512);
        accU[t] = __builtin_amdgcn_wmma_f32_16x16x32_f16(
            false, a, false, BU.v, (short)0, accU[t], false, false);
        accV[t] = __builtin_amdgcn_wmma_f32_16x16x32_f16(
            false, a, false, BV.v, (short)0, accV[t], false, false);
      }
    }
  }

  // ---- epilogue: tanh * sigmoid * w, reduce over all 112 columns ----
  float wl[NT];
#pragma unroll
  for (int t = 0; t < NT; ++t) wl[t] = wpad[t * 16 + lm];

  float ps[8];
#pragma unroll
  for (int r = 0; r < 8; ++r) ps[r] = 0.0f;

#pragma unroll
  for (int t = 0; t < NT; ++t) {
#pragma unroll
    for (int r = 0; r < 8; ++r) {
      float xu = accU[t][r];
      float xv = accV[t][r];
      float e2 = __expf(-2.0f * fabsf(xu));          // stable tanh
      float th = (1.0f - e2) / (1.0f + e2);
      th = (xu < 0.0f) ? -th : th;
      float sg = 1.0f / (1.0f + __expf(-xv));        // sigmoid
      ps[r] += th * sg * wl[t];
    }
  }

  // C/D layout: lane%16 = N; VGPR r = row r (lanes 0-15) / row r+8 (lanes 16-31).
#pragma unroll
  for (int r = 0; r < 8; ++r) {
    float s = ps[r];
    s += __shfl_xor(s, 1);
    s += __shfl_xor(s, 2);
    s += __shfl_xor(s, 4);
    s += __shfl_xor(s, 8);
    ps[r] = s;
  }
  if (lm == 0) {
    const int rbase = row0 + (halfSel ? 8 : 0);
#pragma unroll
    for (int r = 0; r < 8; ++r) alpha[rbase + r] = ps[r];
  }
}

// ---------------------------------------------------------------------------
// Deterministic softmax over B_ROWS (3 passes, tree reductions, no atomics)
// ---------------------------------------------------------------------------
__device__ __forceinline__ float block_reduce(float v, bool is_max) {
  __shared__ float sm[8];
  int tid = threadIdx.x;
#pragma unroll
  for (int off = 16; off > 0; off >>= 1) {
    float o = __shfl_xor(v, off);
    v = is_max ? fmaxf(v, o) : (v + o);
  }
  if ((tid & 31) == 0) sm[tid >> 5] = v;
  __syncthreads();
  float r = (tid < 8) ? sm[tid] : (is_max ? -3.4e38f : 0.0f);
  if (tid < 8) {
#pragma unroll
    for (int off = 4; off > 0; off >>= 1) {
      float o = __shfl_xor(r, off);
      r = is_max ? fmaxf(r, o) : (r + o);
    }
  }
  return r;  // valid in tid==0
}

__global__ __launch_bounds__(256) void reduce_max_kernel(
    const float* __restrict__ x, float* __restrict__ pmax)
{
  float m = -3.4e38f;
  int stride = gridDim.x * 256;
  for (int i = blockIdx.x * 256 + threadIdx.x; i < B_ROWS; i += stride)
    m = fmaxf(m, x[i]);
  float r = block_reduce(m, true);
  if (threadIdx.x == 0) pmax[blockIdx.x] = r;
}

__global__ __launch_bounds__(256) void final_max_kernel(
    const float* __restrict__ pmax, float* __restrict__ gmax)
{
  float r = block_reduce(pmax[threadIdx.x], true);
  if (threadIdx.x == 0) gmax[0] = r;
}

__global__ __launch_bounds__(256) void exp_sum_kernel(
    float* __restrict__ out, const float* __restrict__ gmax,
    float* __restrict__ psum)
{
  const float gm = gmax[0];
  float s = 0.0f;
  int stride = gridDim.x * 256;
  for (int i = blockIdx.x * 256 + threadIdx.x; i < B_ROWS; i += stride) {
    float e = __expf(out[i] - gm);
    out[i] = e;
    s += e;
  }
  float r = block_reduce(s, false);
  if (threadIdx.x == 0) psum[blockIdx.x] = r;
}

__global__ __launch_bounds__(256) void final_sum_kernel(
    const float* __restrict__ psum, float* __restrict__ ginv)
{
  float r = block_reduce(psum[threadIdx.x], false);
  if (threadIdx.x == 0) ginv[0] = 1.0f / r;
}

__global__ __launch_bounds__(256) void scale_kernel(
    float* __restrict__ out, const float* __restrict__ ginv)
{
  const float s = ginv[0];
  int i = blockIdx.x * 256 + threadIdx.x;
  if (i < B_ROWS) out[i] *= s;
}

// ---------------------------------------------------------------------------
extern "C" void kernel_launch(void* const* d_in, const int* in_sizes, int n_in,
                              void* d_out, int out_size, void* d_ws, size_t ws_size,
                              hipStream_t stream)
{
  (void)in_sizes; (void)n_in; (void)out_size; (void)ws_size;
  const float* feat = (const float*)d_in[0];   // (B, C) fp32
  const float* u    = (const float*)d_in[1];   // (P, C) fp32
  const float* v    = (const float*)d_in[2];   // (P, C) fp32
  const float* w    = (const float*)d_in[3];   // (P, 1) fp32
  float* out = (float*)d_out;                  // (B,)   fp32

  // workspace layout (16B-aligned; ~118 KB total)
  char* ws = (char*)d_ws;
  _Float16* uvpack = (_Float16*)(ws + 0);      // 2*57344 = 114688 B
  float*    wpad   = (float*)(ws + 114688);    // 448 B
  float*    gmax   = (float*)(ws + 115200);    // 4 B
  float*    ginv   = (float*)(ws + 115216);    // 4 B
  float*    pmax   = (float*)(ws + 115712);    // 1024 B
  float*    psum   = (float*)(ws + 116736);    // 1024 B

  prep_kernel<<<(PIECES_TOTAL + 255) / 256, 256, 0, stream>>>(u, v, w, uvpack, wpad);
  fused_alpha_kernel<<<B_ROWS / 128, 256, 0, stream>>>(feat, uvpack, wpad, out);
  reduce_max_kernel<<<256, 256, 0, stream>>>(out, pmax);
  final_max_kernel<<<1, 256, 0, stream>>>(pmax, gmax);
  exp_sum_kernel<<<256, 256, 0, stream>>>(out, gmax, psum);
  final_sum_kernel<<<1, 256, 0, stream>>>(psum, ginv);
  scale_kernel<<<B_ROWS / 256, 256, 0, stream>>>(out, ginv);
}
